// EdgeTransformer_90804198572122
// MI455X (gfx1250) — matrix operand order repaired and checked by reading, not compiled, not run on hardware
//
#include <hip/hip_runtime.h>
#include <math.h>
#include <stdint.h>

typedef __attribute__((ext_vector_type(16))) _Float16 v16h;
typedef __attribute__((ext_vector_type(8)))  float    v8f;

#define B_  128
#define C_  100
#define L_  28
#define M_  2
#define E_  64
#define KC_ 10
#define A_  128   // 2E
#define F_  192   // 3E
#define H_  2
#define R_  (B_*C_)   // 12800

// ---------------------------------------------------------------------------
// K1: src branch pooling.  grid = B*C blocks, 128 threads (thread = feature a)
// pooled[b,c,a] = masked temporal mean of [h@Wm+bm  ||  cos(dt*tw+tb)]
// ---------------------------------------------------------------------------
__global__ void k_src_pool(const int* __restrict__ src_ids,
                           const int* __restrict__ cur_ts_p,
                           const float* __restrict__ src_history,
                           const float* __restrict__ src_history_ts,
                           const float* __restrict__ Wm, const float* __restrict__ bm,
                           const float* __restrict__ time_w, const float* __restrict__ time_b,
                           float* __restrict__ pooled)
{
    int bc = blockIdx.x;
    int b = bc / C_, c = bc % C_;
    int a = threadIdx.x;                    // 0..127
    int src = src_ids[b];
    float cur = (float)(*cur_ts_p);
    const float* tsp = src_history_ts + ((long)src * C_ + c) * L_;
    const float* hp  = src_history + (((long)src * C_ + c) * L_) * M_;
    float w0 = 0.f, w1 = 0.f, bb = 0.f, tw = 0.f, tb = 0.f;
    if (a < E_) { w0 = Wm[a]; w1 = Wm[E_ + a]; bb = bm[a]; }
    else        { tw = time_w[a - E_]; tb = time_b[a - E_]; }
    float acc = 0.f, cnt = 0.f;
    for (int l = 0; l < L_; ++l) {
        float ts = tsp[l];
        float m  = (ts > 0.f) ? 1.f : 0.f;
        float val;
        if (a < E_) val = hp[l*M_] * w0 + hp[l*M_+1] * w1 + bb;
        else        val = cosf((cur - ts) * tw + tb);
        acc += m * val;
        cnt += m;
    }
    pooled[(long)bc * A_ + a] = acc / fmaxf(cnt, 1.f);
}

// ---------------------------------------------------------------------------
// K2: softmax routing over K clusters + weighted cluster aggregation + K_emb.
// grid = B blocks, 128 threads.
// ---------------------------------------------------------------------------
__global__ void k_route(const float* __restrict__ pooled,
                        const float* __restrict__ Wq, const float* __restrict__ bq,
                        const float* __restrict__ K_emb,
                        float* __restrict__ clus)
{
    __shared__ float probsL[C_ * KC_];
    int b = blockIdx.x, t = threadIdx.x;
    const float* pb = pooled + (long)b * C_ * A_;
    if (t < C_) {
        float lg[KC_];
        #pragma unroll
        for (int k = 0; k < KC_; ++k) lg[k] = bq[k];
        const float* pr = pb + t * A_;
        for (int a = 0; a < A_; ++a) {
            float p = pr[a];
            const float* wq = Wq + a * KC_;
            #pragma unroll
            for (int k = 0; k < KC_; ++k) lg[k] += p * wq[k];
        }
        float mx = lg[0];
        #pragma unroll
        for (int k = 1; k < KC_; ++k) mx = fmaxf(mx, lg[k]);
        float s = 0.f;
        #pragma unroll
        for (int k = 0; k < KC_; ++k) { lg[k] = expf(lg[k] - mx); s += lg[k]; }
        float inv = 1.f / s;
        #pragma unroll
        for (int k = 0; k < KC_; ++k) probsL[t*KC_ + k] = lg[k] * inv;
    }
    __syncthreads();
    float cl[KC_], ps[KC_];
    #pragma unroll
    for (int k = 0; k < KC_; ++k) { cl[k] = 0.f; ps[k] = 0.f; }
    for (int c = 0; c < C_; ++c) {
        float p = pb[c*A_ + t];
        #pragma unroll
        for (int k = 0; k < KC_; ++k) {
            float q = probsL[c*KC_ + k];
            cl[k] += q * p;
            ps[k] += q;
        }
    }
    #pragma unroll
    for (int k = 0; k < KC_; ++k) {
        float v = cl[k] / fmaxf(ps[k], 1e-6f);
        if (t < E_) v += K_emb[k*E_ + t];
        clus[((long)b*KC_ + k)*A_ + t] = v;
    }
}

// ---------------------------------------------------------------------------
// K3: per-batch MHA over 10 clusters + SiLU FFN + layernorm + mean.
// grid = B blocks, 128 threads (thread = feature dim).
// ---------------------------------------------------------------------------
__global__ void k_mha(const float* __restrict__ clus_g,
                      const float* __restrict__ Wqkv, const float* __restrict__ bqkv,
                      const float* __restrict__ Wo, const float* __restrict__ bo,
                      const float* __restrict__ Ws1, const float* __restrict__ bs1,
                      const float* __restrict__ Ws2, const float* __restrict__ bs2,
                      const float* __restrict__ sn_g, const float* __restrict__ sn_b,
                      float* __restrict__ src_feat)
{
    __shared__ float xL[KC_][A_];
    __shared__ float qL[KC_][A_], kL[KC_][A_], vL[KC_][A_];
    __shared__ float sL[H_][KC_][KC_];
    __shared__ float oL[KC_][A_];
    __shared__ float yL[KC_][A_];
    __shared__ float hL[KC_][2*A_];
    __shared__ float zL[KC_][A_];
    __shared__ float muL[KC_], rsL[KC_];
    int b = blockIdx.x, t = threadIdx.x;
    for (int r = 0; r < KC_; ++r) xL[r][t] = clus_g[((long)b*KC_ + r)*A_ + t];
    __syncthreads();
    // QKV projection: thread t owns columns t / 128+t / 256+t
    for (int r = 0; r < KC_; ++r) {
        float aq = bqkv[t], ak = bqkv[A_ + t], av = bqkv[2*A_ + t];
        for (int a = 0; a < A_; ++a) {
            float x = xL[r][a];
            const float* w = Wqkv + a * 3 * A_;
            aq += x * w[t];
            ak += x * w[A_ + t];
            av += x * w[2*A_ + t];
        }
        qL[r][t] = aq; kL[r][t] = ak; vL[r][t] = av;
    }
    __syncthreads();
    // attention scores, 2 heads, dh=64, scale 1/8
    if (t < KC_*KC_) {
        int i = t / KC_, j = t % KC_;
        #pragma unroll
        for (int h = 0; h < H_; ++h) {
            float d = 0.f;
            for (int e = 0; e < E_; ++e) d += qL[i][h*E_ + e] * kL[j][h*E_ + e];
            sL[h][i][j] = d * 0.125f;
        }
    }
    __syncthreads();
    if (t < H_*KC_) {
        int h = t / KC_, i = t % KC_;
        float mx = sL[h][i][0];
        for (int j = 1; j < KC_; ++j) mx = fmaxf(mx, sL[h][i][j]);
        float s = 0.f;
        for (int j = 0; j < KC_; ++j) { float e = expf(sL[h][i][j] - mx); sL[h][i][j] = e; s += e; }
        float inv = 1.f / s;
        for (int j = 0; j < KC_; ++j) sL[h][i][j] *= inv;
    }
    __syncthreads();
    {
        int h = t >> 6;                 // head of this output column
        for (int r = 0; r < KC_; ++r) {
            float acc = 0.f;
            #pragma unroll
            for (int j = 0; j < KC_; ++j) acc += sL[h][r][j] * vL[j][t];
            oL[r][t] = acc;
        }
    }
    __syncthreads();
    // output projection + residual
    for (int r = 0; r < KC_; ++r) {
        float acc = bo[t];
        for (int a = 0; a < A_; ++a) acc += oL[r][a] * Wo[a*A_ + t];
        yL[r][t] = xL[r][t] + acc;
    }
    __syncthreads();
    // FFN: 128 -> 256 (SiLU) -> 128, residual
    for (int r = 0; r < KC_; ++r) {
        float a1 = bs1[t], a2 = bs1[A_ + t];
        for (int a = 0; a < A_; ++a) {
            float x = yL[r][a];
            a1 += x * Ws1[a*2*A_ + t];
            a2 += x * Ws1[a*2*A_ + A_ + t];
        }
        hL[r][t]      = a1 / (1.f + expf(-a1));
        hL[r][A_ + t] = a2 / (1.f + expf(-a2));
    }
    __syncthreads();
    for (int r = 0; r < KC_; ++r) {
        float acc = bs2[t];
        for (int j = 0; j < 2*A_; ++j) acc += hL[r][j] * Ws2[j*A_ + t];
        zL[r][t] = yL[r][t] + acc;
    }
    __syncthreads();
    if (t < KC_) {
        float s = 0.f;
        for (int a = 0; a < A_; ++a) s += zL[t][a];
        float mu = s / A_;
        float v = 0.f;
        for (int a = 0; a < A_; ++a) { float d = zL[t][a] - mu; v += d*d; }
        muL[t] = mu;
        rsL[t] = rsqrtf(v / A_ + 1e-5f);
    }
    __syncthreads();
    float sf = 0.f;
    for (int r = 0; r < KC_; ++r)
        sf += (zL[r][t] - muL[r]) * rsL[r] * sn_g[t] + sn_b[t];
    src_feat[(long)b*A_ + t] = sf / (float)KC_;
}

// ---------------------------------------------------------------------------
// K4: dst branch pooling + d_mlp3.  grid = C blocks, 128 threads.
// ---------------------------------------------------------------------------
__global__ void k_dst(const int* __restrict__ cur_ts_p,
                      const float* __restrict__ dst_history,
                      const float* __restrict__ dst_history_ts,
                      const float* __restrict__ Wdm, const float* __restrict__ bdm,
                      const float* __restrict__ time_w, const float* __restrict__ time_b,
                      const float* __restrict__ Wd1, const float* __restrict__ bd1,
                      const float* __restrict__ Wd2, const float* __restrict__ bd2,
                      float* __restrict__ dfeat)
{
    __shared__ float dpool[A_];
    __shared__ float hid[A_];
    int c = blockIdx.x, a = threadIdx.x;
    float cur = (float)(*cur_ts_p);
    const float* tsp = dst_history_ts + c * L_;
    const float* hp  = dst_history + c * L_ * M_;
    float w0 = 0.f, w1 = 0.f, bb = 0.f, tw = 0.f, tb = 0.f;
    if (a < E_) { w0 = Wdm[a]; w1 = Wdm[E_ + a]; bb = bdm[a]; }
    else        { tw = time_w[a - E_]; tb = time_b[a - E_]; }
    float acc = 0.f, cnt = 0.f;
    for (int l = 0; l < L_; ++l) {
        float ts = tsp[l];
        float m  = (ts > 0.f) ? 1.f : 0.f;
        float val;
        if (a < E_) val = hp[l*M_] * w0 + hp[l*M_+1] * w1 + bb;
        else        val = cosf((cur - ts) * tw + tb);
        acc += m * val;
        cnt += m;
    }
    dpool[a] = acc / fmaxf(cnt, 1.f);
    __syncthreads();
    float h1 = bd1[a];
    for (int j = 0; j < A_; ++j) h1 += dpool[j] * Wd1[j*A_ + a];
    hid[a] = h1 / (1.f + expf(-h1));
    __syncthreads();
    if (a < E_) {
        float d = bd2[a];
        for (int j = 0; j < A_; ++j) d += hid[j] * Wd2[j*E_ + a];
        dfeat[c*E_ + a] = d;
    }
}

// ---------------------------------------------------------------------------
// K5: repack fp32 weight [Kdim,Ndim] into f16 WMMA B-fragments.
// Fragment (nt,ks): 32 lanes x 16 halves contiguous.  B k order used here (A matches):
// VGPRv lanes0-15 hold K=2v,2v+1; lanes16-31 hold K=2v+16,2v+17; N=lane&15.
// ---------------------------------------------------------------------------
__global__ void k_pack_b(const float* __restrict__ W, _Float16* __restrict__ out,
                         int Kdim, int Ndim)
{
    int KS = Kdim >> 5;
    int total = Kdim * Ndim;
    int o = blockIdx.x * blockDim.x + threadIdx.x;
    if (o >= total) return;
    int frag = o >> 9;          // 512 halves per fragment
    int e5   = o & 511;
    int lane = e5 >> 4;
    int e    = e5 & 15;
    int v    = e >> 1, p = e & 1;
    int nt = frag / KS, ks = frag % KS;
    int hi = (lane >= 16) ? 16 : 0;
    int Krow = ks*32 + 2*v + p + hi;
    int n    = nt*16 + (lane & 15);
    out[o] = (_Float16)W[Krow * Ndim + n];
}

// ---------------------------------------------------------------------------
// K6: fused final MLP with WMMA.
// 800 blocks x 128 threads (4 waves).  Tile = 16 fused rows of f=[src||dst].
// GEMM1: [16,192]x[192,384] + bias + SiLU (v_wmma_f32_16x16x32_f16)
// GEMM2: [16,384]x[384,192] + bias + residual, then LN + Wp dot.
// ---------------------------------------------------------------------------
__global__ void __launch_bounds__(128)
k_final(const float* __restrict__ src_feat,
        const float* __restrict__ dfeat,
        const _Float16* __restrict__ w1h, const float* __restrict__ bf1,
        const _Float16* __restrict__ w2h, const float* __restrict__ bf2,
        const float* __restrict__ ln_g, const float* __restrict__ ln_b,
        const float* __restrict__ Wp, const float* __restrict__ bp,
        float* __restrict__ out)
{
    __shared__ float    fF32[16][F_];       // residual copy
    __shared__ _Float16 fF16[16][F_];       // GEMM1 A source
    __shared__ _Float16 h1F16[16][2*F_];    // GEMM2 A source
    __shared__ float    yF32[16][F_];       // pre-LN result

    int tid  = threadIdx.x;
    int wave = tid >> 5, lane = tid & 31;
    int row16 = lane & 15;
    int hi = (lane >> 4) & 1;
    long r0 = (long)blockIdx.x * 16;

    // build fused input tile f = [src_feat[b] || dfeat[c]]
    for (int idx = tid; idx < 16*F_; idx += 128) {
        int rr = idx / F_, col = idx % F_;
        long grow = r0 + rr;
        int b = (int)(grow / C_), c = (int)(grow % C_);
        float val = (col < A_) ? src_feat[(long)b*A_ + col]
                               : dfeat[c*E_ + (col - A_)];
        fF32[rr][col] = val;
        fF16[rr][col] = (_Float16)val;
    }
    __syncthreads();

    // ---- GEMM1 ----  A fragments hoisted (16-bit A 16x32 lane layout)
    v16h aF[6];
    #pragma unroll
    for (int ks = 0; ks < 6; ++ks) {
        #pragma unroll
        for (int v = 0; v < 8; ++v) {
            int kb = 2*v + (hi ? 16 : 0);
            aF[ks][2*v]   = fF16[row16][ks*32 + kb];
            aF[ks][2*v+1] = fF16[row16][ks*32 + kb + 1];
        }
    }
    const v16h* w1v = (const v16h*)w1h;
    #pragma unroll
    for (int i = 0; i < 6; ++i) {          // 24 N-tiles / 4 waves
        int nt = wave + 4*i;
        float bv = bf1[nt*16 + row16];
        v8f acc;
        #pragma unroll
        for (int r = 0; r < 8; ++r) acc[r] = bv;
        #pragma unroll
        for (int ks = 0; ks < 6; ++ks) {
            v16h bF = w1v[(nt*6 + ks)*32 + lane];
            acc = __builtin_amdgcn_wmma_f32_16x16x32_f16(false, aF[ks], false, bF,
                                                         (short)0, acc, false, false);
        }
        #pragma unroll
        for (int r = 0; r < 8; ++r) {
            float x = acc[r];
            float s = x / (1.f + expf(-x));             // SiLU
            h1F16[r + (hi ? 8 : 0)][nt*16 + row16] = (_Float16)s;
        }
    }
    __syncthreads();

    // ---- GEMM2 + residual ----
    const v16h* w2v = (const v16h*)w2h;
    #pragma unroll
    for (int i = 0; i < 3; ++i) {          // 12 N-tiles / 4 waves
        int nt = wave + 4*i;
        float bv = bf2[nt*16 + row16];
        v8f acc;
        #pragma unroll
        for (int r = 0; r < 8; ++r) acc[r] = bv;
        for (int ks = 0; ks < 12; ++ks) {
            v16h aG;
            #pragma unroll
            for (int v = 0; v < 8; ++v) {
                int kb = 2*v + (hi ? 16 : 0);
                aG[2*v]   = h1F16[row16][ks*32 + kb];
                aG[2*v+1] = h1F16[row16][ks*32 + kb + 1];
            }
            v16h bF = w2v[(nt*12 + ks)*32 + lane];
            acc = __builtin_amdgcn_wmma_f32_16x16x32_f16(false, aG, false, bF,
                                                         (short)0, acc, false, false);
        }
        #pragma unroll
        for (int r = 0; r < 8; ++r) {
            int rr  = r + (hi ? 8 : 0);
            int col = nt*16 + row16;
            yF32[rr][col] = acc[r] + fF32[rr][col];     // residual
        }
    }
    __syncthreads();

    // ---- LayerNorm + Wp dot ----
    if (tid < 16) {
        int rr = tid;
        float s = 0.f;
        for (int col = 0; col < F_; ++col) s += yF32[rr][col];
        float mu = s / F_;
        float var = 0.f;
        for (int col = 0; col < F_; ++col) { float d = yF32[rr][col] - mu; var += d*d; }
        float rs = rsqrtf(var / F_ + 1e-5f);
        float score = bp[0];
        for (int col = 0; col < F_; ++col) {
            float fn = (yF32[rr][col] - mu) * rs * ln_g[col] + ln_b[col];
            score += fn * Wp[col];
        }
        out[r0 + rr] = score;
    }
}

// ---------------------------------------------------------------------------
extern "C" void kernel_launch(void* const* d_in, const int* in_sizes, int n_in,
                              void* d_out, int out_size, void* d_ws, size_t ws_size,
                              hipStream_t stream)
{
    (void)in_sizes; (void)n_in; (void)out_size; (void)ws_size;
    const int*   src_ids        = (const int*)d_in[0];
    const int*   cur_ts         = (const int*)d_in[1];
    const float* src_history    = (const float*)d_in[2];
    const float* src_history_ts = (const float*)d_in[3];
    const float* dst_history    = (const float*)d_in[4];
    const float* dst_history_ts = (const float*)d_in[5];
    const float* Wm   = (const float*)d_in[6];
    const float* bm   = (const float*)d_in[7];
    const float* Wdm  = (const float*)d_in[8];
    const float* bdm  = (const float*)d_in[9];
    const float* time_w = (const float*)d_in[10];
    const float* time_b = (const float*)d_in[11];
    const float* Wq   = (const float*)d_in[12];
    const float* bq   = (const float*)d_in[13];
    const float* K_emb= (const float*)d_in[14];
    const float* Wqkv = (const float*)d_in[15];
    const float* bqkv = (const float*)d_in[16];
    const float* Wo   = (const float*)d_in[17];
    const float* bo   = (const float*)d_in[18];
    const float* Ws1  = (const float*)d_in[19];
    const float* bs1  = (const float*)d_in[20];
    const float* Ws2  = (const float*)d_in[21];
    const float* bs2  = (const float*)d_in[22];
    const float* sn_g = (const float*)d_in[23];
    const float* sn_b = (const float*)d_in[24];
    const float* Wd1  = (const float*)d_in[25];
    const float* bd1  = (const float*)d_in[26];
    const float* Wd2  = (const float*)d_in[27];
    const float* bd2  = (const float*)d_in[28];
    const float* Wf1  = (const float*)d_in[29];
    const float* bf1  = (const float*)d_in[30];
    const float* Wf2  = (const float*)d_in[31];
    const float* bf2  = (const float*)d_in[32];
    const float* ln_g = (const float*)d_in[33];
    const float* ln_b = (const float*)d_in[34];
    const float* Wp   = (const float*)d_in[35];
    const float* bp   = (const float*)d_in[36];

    // workspace carve-out (~7.6 MB)
    char* ws = (char*)d_ws;
    float* pooled   = (float*)ws;  ws += sizeof(float)*(size_t)B_*C_*A_;
    float* clus     = (float*)ws;  ws += sizeof(float)*(size_t)B_*KC_*A_;
    float* src_feat = (float*)ws;  ws += sizeof(float)*(size_t)B_*A_;
    float* dfeat    = (float*)ws;  ws += sizeof(float)*(size_t)C_*E_;
    uintptr_t u = (uintptr_t)ws; u = (u + 31) & ~(uintptr_t)31;
    _Float16* w1h = (_Float16*)u;  u += sizeof(_Float16)*(size_t)F_*2*F_;
    u = (u + 31) & ~(uintptr_t)31;
    _Float16* w2h = (_Float16*)u;

    k_pack_b<<<(F_*2*F_ + 255)/256, 256, 0, stream>>>(Wf1, w1h, F_, 2*F_);
    k_pack_b<<<(2*F_*F_ + 255)/256, 256, 0, stream>>>(Wf2, w2h, 2*F_, F_);
    k_src_pool<<<B_*C_, 128, 0, stream>>>(src_ids, cur_ts, src_history, src_history_ts,
                                          Wm, bm, time_w, time_b, pooled);
    k_dst<<<C_, 128, 0, stream>>>(cur_ts, dst_history, dst_history_ts, Wdm, bdm,
                                  time_w, time_b, Wd1, bd1, Wd2, bd2, dfeat);
    k_route<<<B_, 128, 0, stream>>>(pooled, Wq, bq, K_emb, clus);
    k_mha<<<B_, 128, 0, stream>>>(clus, Wqkv, bqkv, Wo, bo, Ws1, bs1, Ws2, bs2,
                                  sn_g, sn_b, src_feat);
    k_final<<<R_/16, 128, 0, stream>>>(src_feat, dfeat, w1h, bf1, w2h, bf2,
                                       ln_g, ln_b, Wp, bp, (float*)d_out);
}
